// EMPNNModel_52948356825654
// MI455X (gfx1250) — matrix operand order repaired and checked by reading, not compiled, not run on hardware
//
#include <hip/hip_runtime.h>
#include <hip/hip_bf16.h>
#include <math.h>

// ---------------------------------------------------------------------------
// Types for CDNA5 WMMA (wave32): 16x16x32 bf16 -> f32
// ---------------------------------------------------------------------------
typedef __bf16 v16bf __attribute__((ext_vector_type(16)));
typedef __bf16 v8bf  __attribute__((ext_vector_type(8)));
typedef float  v8f   __attribute__((ext_vector_type(8)));

__device__ __forceinline__ __bf16 f2bf(float f) {
    unsigned u = __builtin_bit_cast(unsigned, f);
    u = (u + 0x7FFFu + ((u >> 16) & 1u)) >> 16;           // round-to-nearest-even
    return __builtin_bit_cast(__bf16, (unsigned short)u);
}

__device__ __forceinline__ float silu(float x) { return x / (1.f + __expf(-x)); }

// ---------------------------------------------------------------------------
// WMMA GEMM (pre-packed bf16 operands):
//   C[M x N] = act(A[M x K] @ W[K x N] + bias)
//   A  : bf16 row-major (M x K)
//   Wt : bf16 row-major (N x K)  == W transposed
//   K % 32 == 0, N % 64 == 0, M arbitrary (tail masked on store, clamped load)
//
// Block = 128 threads (4 waves). Wave w owns a 64-row x 16-col strip:
//   4 accumulators (16x16 each), B fragment reused across the 4 M-tiles.
// Fragment loads are pure b128 vector loads:
//   A frag  = A[row][k0+8*hh .. +7]  and  A[row][k0+16+8*hh .. +7]
//   B frag  = Wt[ncol][k0+8*hh .. +7]  and  Wt[ncol][k0+16+8*hh .. +7]
// act: 0 = none, 1 = SiLU.  C (f32) and Cbf (bf16) outputs each optional.
// ---------------------------------------------------------------------------
__global__ void __launch_bounds__(128)
gemm_wmma_bf16(const __bf16* __restrict__ A, const __bf16* __restrict__ Wt,
               const float* __restrict__ bias,
               float* __restrict__ C, __bf16* __restrict__ Cbf,
               int M, int K, int N, int act)
{
    const int tid  = threadIdx.x;
    const int wave = tid >> 5;
    const int lane = tid & 31;
    const int m    = lane & 15;
    const int hh   = lane >> 4;

    const int row0 = blockIdx.x * 64;
    const int ncol = blockIdx.y * 64 + wave * 16 + m;

    v8f acc0 = {0.f,0.f,0.f,0.f,0.f,0.f,0.f,0.f};
    v8f acc1 = acc0, acc2 = acc0, acc3 = acc0;

    const __bf16* wrow = Wt + (size_t)ncol * K;

    // clamped row pointers for the 4 M-subtiles (OOB rows read row M-1)
    int r0 = row0 +  0 + m; r0 = r0 < M ? r0 : M - 1;
    int r1 = row0 + 16 + m; r1 = r1 < M ? r1 : M - 1;
    int r2 = row0 + 32 + m; r2 = r2 < M ? r2 : M - 1;
    int r3 = row0 + 48 + m; r3 = r3 < M ? r3 : M - 1;
    const __bf16* a0 = A + (size_t)r0 * K;
    const __bf16* a1 = A + (size_t)r1 * K;
    const __bf16* a2 = A + (size_t)r2 * K;
    const __bf16* a3 = A + (size_t)r3 * K;

    for (int k0 = 0; k0 < K; k0 += 32) {
        // B fragment: two 8-element bf16 runs (two b128 loads), same k order as A
        v8bf blo = *(const v8bf*)(wrow + k0 + 8 * hh);
        v8bf bhi = *(const v8bf*)(wrow + k0 + 16 + 8 * hh);
        v16bf bf_ = __builtin_shufflevector(blo, bhi,
                        0,1,2,3,4,5,6,7,8,9,10,11,12,13,14,15);

        // A fragments: two b128 loads each
        #define LOAD_A(ap)                                                     \
            __builtin_shufflevector(*(const v8bf*)((ap) + k0 + 8 * hh),        \
                                    *(const v8bf*)((ap) + k0 + 16 + 8 * hh),   \
                                    0,1,2,3,4,5,6,7,8,9,10,11,12,13,14,15)
        v16bf af0 = LOAD_A(a0);
        v16bf af1 = LOAD_A(a1);
        v16bf af2 = LOAD_A(a2);
        v16bf af3 = LOAD_A(a3);
        #undef LOAD_A

        acc0 = __builtin_amdgcn_wmma_f32_16x16x32_bf16(false, af0, false, bf_,
                                                       (short)0, acc0, false, false);
        acc1 = __builtin_amdgcn_wmma_f32_16x16x32_bf16(false, af1, false, bf_,
                                                       (short)0, acc1, false, false);
        acc2 = __builtin_amdgcn_wmma_f32_16x16x32_bf16(false, af2, false, bf_,
                                                       (short)0, acc2, false, false);
        acc3 = __builtin_amdgcn_wmma_f32_16x16x32_bf16(false, af3, false, bf_,
                                                       (short)0, acc3, false, false);
    }

    // Epilogue. C/D layout: VGPR r -> row = r + 8*hh, col = lane%16.
    float bn = bias ? bias[ncol] : 0.f;
    #pragma unroll
    for (int t = 0; t < 4; ++t) {
        v8f accv = t == 0 ? acc0 : t == 1 ? acc1 : t == 2 ? acc2 : acc3;
        #pragma unroll
        for (int r = 0; r < 8; ++r) {
            int row = row0 + t * 16 + r + 8 * hh;
            if (row < M) {
                float val = accv[r] + bn;
                if (act) val = silu(val);
                if (C)   C  [(size_t)row * N + ncol] = val;
                if (Cbf) Cbf[(size_t)row * N + ncol] = f2bf(val);
            }
        }
    }
}

// ---------------------------------------------------------------------------
// Packing kernels
// ---------------------------------------------------------------------------
__global__ void pack_bf16(const float* __restrict__ in, __bf16* __restrict__ out,
                          long len)
{
    long i = (long)blockIdx.x * blockDim.x + threadIdx.x;
    if (i < len) out[i] = f2bf(in[i]);
}

// W (K x N) f32 -> Wt (N x K) bf16
__global__ void pack_w_t(const float* __restrict__ W, __bf16* __restrict__ Wt,
                         int K, int N)
{
    int idx = blockIdx.x * blockDim.x + threadIdx.x;
    if (idx >= K * N) return;
    int n_ = idx / K, k = idx - n_ * K;
    Wt[idx] = f2bf(W[(size_t)k * N + n_]);
}

// ---------------------------------------------------------------------------
// Node embedding (one block per node)
// ---------------------------------------------------------------------------
__global__ void __launch_bounds__(128)
embed_kernel(const float* __restrict__ na, const float* __restrict__ x,
             const float* __restrict__ We,
             const float* __restrict__ W1e, const float* __restrict__ b1e,
             const float* __restrict__ W2e, const float* __restrict__ b2e,
             float* __restrict__ s, float* __restrict__ v, int n)
{
    __shared__ float ctx[130];
    __shared__ float hid[128];
    int i = blockIdx.x;
    if (i >= n) return;
    int t = threadIdx.x;

    float na0 = na[2 * i], na1 = na[2 * i + 1];
    if (t < 2) ctx[t] = x[2 * i + t];
    float r = sqrtf(na0 * na0 + na1 * na1);
    ctx[2 + t] = fabsf(We[t]) * r;
    __syncthreads();

    float acc = b1e[t];
    for (int k = 0; k < 130; ++k) acc += ctx[k] * W1e[k * 128 + t];
    hid[t] = silu(acc);
    __syncthreads();

    float o0 = b2e[t], o1 = b2e[128 + t];
    for (int k = 0; k < 128; ++k) {
        float h = hid[k];
        o0 += h * W2e[k * 256 + t];
        o1 += h * W2e[k * 256 + 128 + t];
    }
    s[(size_t)i * 128 + t] = o0;
    float wv = We[128 + t];
    v[(size_t)i * 256 + t]       = o1 * na0 * wv;
    v[(size_t)i * 256 + 128 + t] = o1 * na1 * wv;
}

// ---------------------------------------------------------------------------
// Edge preprocessing: d_ij, dir_ij
// ---------------------------------------------------------------------------
__global__ void edge_pre(const float* __restrict__ ea,
                         float* __restrict__ dij, float* __restrict__ dire, int E)
{
    int e = blockIdx.x * blockDim.x + threadIdx.x;
    if (e >= E) return;
    float a0 = ea[2 * e], a1 = ea[2 * e + 1];
    float d = sqrtf(a0 * a0 + a1 * a1) + 1e-8f;
    dij[e] = d;
    dire[2 * e] = a0 / d;
    dire[2 * e + 1] = a1 / d;
}

// ---------------------------------------------------------------------------
// Edge message + scatter-add (one block per edge, 128 threads = feature dim)
// ---------------------------------------------------------------------------
__global__ void __launch_bounds__(128)
edge_msg(const float* __restrict__ phi,     // (n, 384)
         const float* __restrict__ Wfilt, const float* __restrict__ bfilt,
         const float* __restrict__ dij, const float* __restrict__ dire,
         const float* __restrict__ vold,    // (n, 2, 128) snapshot
         const int* __restrict__ ej, const int* __restrict__ ei,
         float* __restrict__ s, float* __restrict__ v, int E)
{
    int e = blockIdx.x;
    if (e >= E) return;
    int t = threadIdx.x;

    int j = ej[e], i = ei[e];
    float d  = dij[e];
    float d0 = dire[2 * e], d1 = dire[2 * e + 1];

    const float* pj = phi + (size_t)j * 384;
    __builtin_prefetch(pj, 0, 1);                    // global_prefetch_b8
    const float* vj = vold + (size_t)j * 256;

    float wf0 = d * Wfilt[t]       + bfilt[t];
    float wf1 = d * Wfilt[128 + t] + bfilt[128 + t];
    float wf2 = d * Wfilt[256 + t] + bfilt[256 + t];

    float ds  = pj[t]       * wf0;
    float dvs = pj[128 + t] * wf1;
    float dvv = pj[256 + t] * wf2;

    atomicAdd(&s[(size_t)i * 128 + t],       ds);
    atomicAdd(&v[(size_t)i * 256 + t],       dvs * vj[t]       + dvv * d0);
    atomicAdd(&v[(size_t)i * 256 + 128 + t], dvs * vj[128 + t] + dvv * d1);
}

// ---------------------------------------------------------------------------
// ctx build (bf16 out):  ctx[i] = [ s[i], || V2[i,:,:] ||_channels ]
// ---------------------------------------------------------------------------
__global__ void __launch_bounds__(128)
ctx_kernel(const float* __restrict__ s, const float* __restrict__ V2,
           __bf16* __restrict__ ctxbf, int n)
{
    int i = blockIdx.x;
    if (i >= n) return;
    int t = threadIdx.x;
    float w0 = V2[(size_t)i * 256 + t];
    float w1 = V2[(size_t)i * 256 + 128 + t];
    ctxbf[(size_t)i * 256 + t]       = f2bf(s[(size_t)i * 128 + t]);
    ctxbf[(size_t)i * 256 + 128 + t] = f2bf(sqrtf(w0 * w0 + w1 * w1));
}

// ---------------------------------------------------------------------------
// Gated update:  v += a_vv*Uv ;  s += a_ss + a_sv * <Uv, V2>_channels
// ---------------------------------------------------------------------------
__global__ void __launch_bounds__(128)
update_kernel(const float* __restrict__ a,   // (n, 384) = [a_vv | a_sv | a_ss]
              const float* __restrict__ U, const float* __restrict__ V2,
              float* __restrict__ s, float* __restrict__ v, int n)
{
    int i = blockIdx.x;
    if (i >= n) return;
    int t = threadIdx.x;
    float avv = a[(size_t)i * 384 + t];
    float asv = a[(size_t)i * 384 + 128 + t];
    float ass = a[(size_t)i * 384 + 256 + t];
    float u0 = U[(size_t)i * 256 + t],  u1 = U[(size_t)i * 256 + 128 + t];
    float w0 = V2[(size_t)i * 256 + t], w1 = V2[(size_t)i * 256 + 128 + t];
    v[(size_t)i * 256 + t]       += avv * u0;
    v[(size_t)i * 256 + 128 + t] += avv * u1;
    s[(size_t)i * 128 + t]       += ass + asv * (u0 * w0 + u1 * w1);
}

// ---------------------------------------------------------------------------
// FC head on node 0
// ---------------------------------------------------------------------------
__global__ void __launch_bounds__(128)
head_kernel(const float* __restrict__ s,
            const float* __restrict__ Wout, const float* __restrict__ bout,
            const float* __restrict__ Wfin, const float* __restrict__ bfin,
            float* __restrict__ out)
{
    __shared__ float h[128];
    __shared__ float nh[128];
    __shared__ float red[128];
    int t = threadIdx.x;
    h[t] = s[t];
    __syncthreads();
    for (int l = 0; l < 6; ++l) {
        const float* W = Wout + (size_t)l * 128 * 128;
        float acc = bout[l * 128 + t];
        for (int k = 0; k < 128; ++k) acc += h[k] * W[k * 128 + t];
        nh[t] = tanhf(acc);
        __syncthreads();
        h[t] = nh[t];
        __syncthreads();
    }
    red[t] = h[t] * Wfin[t];
    __syncthreads();
    for (int off = 64; off > 0; off >>= 1) {
        if (t < off) red[t] += red[t + off];
        __syncthreads();
    }
    if (t == 0) out[0] = red[0] + bfin[0];
}

// ---------------------------------------------------------------------------
// Host-side orchestration
// ---------------------------------------------------------------------------
extern "C" void kernel_launch(void* const* d_in, const int* in_sizes, int n_in,
                              void* d_out, int out_size, void* d_ws, size_t ws_size,
                              hipStream_t stream)
{
    (void)n_in; (void)out_size; (void)ws_size;
    const int H = 128;
    const int n = in_sizes[0] / 2;
    const int E = in_sizes[2] / 2;

    const float* node_attr = (const float*)d_in[0];
    const float* x         = (const float*)d_in[1];
    const float* edge_attr = (const float*)d_in[2];
    const int*   edge_idx  = (const int*)  d_in[3];
    const float* Wv_embed  = (const float*)d_in[4];
    const float* W1e       = (const float*)d_in[5];
    const float* b1e       = (const float*)d_in[6];
    const float* W2e       = (const float*)d_in[7];
    const float* b2e       = (const float*)d_in[8];
    const float* Wphi1     = (const float*)d_in[9];
    const float* bphi1     = (const float*)d_in[10];
    const float* Wphi2     = (const float*)d_in[11];
    const float* bphi2     = (const float*)d_in[12];
    const float* Wfilt     = (const float*)d_in[13];
    const float* bfilt     = (const float*)d_in[14];
    const float* WU        = (const float*)d_in[15];
    const float* WV        = (const float*)d_in[16];
    const float* W1u       = (const float*)d_in[17];
    const float* b1u       = (const float*)d_in[18];
    const float* W2u       = (const float*)d_in[19];
    const float* b2u       = (const float*)d_in[20];
    const float* Wout      = (const float*)d_in[21];
    const float* bout      = (const float*)d_in[22];
    const float* Wfin      = (const float*)d_in[23];
    const float* bfin      = (const float*)d_in[24];

    const int* ej = edge_idx;
    const int* ei = edge_idx + E;

    // ---------------- workspace layout ----------------
    float* s    = (float*)d_ws;               // n*128
    float* v    = s    + (size_t)n * 128;     // n*256
    float* vold = v    + (size_t)n * 256;     // n*256
    float* U    = vold + (size_t)n * 256;     // n*256
    float* V2   = U    + (size_t)n * 256;     // n*256
    float* big  = V2   + (size_t)n * 256;     // n*384
    float* dij  = big  + (size_t)n * 384;     // E
    float* dire = dij  + (size_t)E;           // 2E
    __bf16* sbf   = (__bf16*)(dire + (size_t)2 * E); // n*128
    __bf16* hidbf = sbf   + (size_t)n * 128;         // n*128
    __bf16* ctxbf = hidbf + (size_t)n * 128;         // n*256
    __bf16* vbf   = ctxbf + (size_t)n * 256;         // n*256
    __bf16* wbf   = vbf   + (size_t)n * 256;         // 4 * 180224 bf16

    // per-layer transposed bf16 weight offsets (elements)
    const size_t LSTRIDE = 180224;
    const size_t OFF_PHI1 = 0;        // 128x128   -> 16384
    const size_t OFF_PHI2 = 16384;    // 384x128   -> 49152
    const size_t OFF_WU   = 65536;    // 128x128
    const size_t OFF_WV   = 81920;    // 128x128
    const size_t OFF_W1U  = 98304;    // 128x256   -> 32768
    const size_t OFF_W2U  = 131072;   // 384x128   -> 49152

    // ---------------- weight packing (bf16, transposed) ----------------
    for (int l = 0; l < 4; ++l) {
        __bf16* wl = wbf + (size_t)l * LSTRIDE;
        pack_w_t<<<(128*128 + 255)/256, 256, 0, stream>>>(
            Wphi1 + (size_t)l*H*H,     wl + OFF_PHI1, H,   H);
        pack_w_t<<<(128*384 + 255)/256, 256, 0, stream>>>(
            Wphi2 + (size_t)l*H*3*H,   wl + OFF_PHI2, H,   3*H);
        pack_w_t<<<(128*128 + 255)/256, 256, 0, stream>>>(
            WU    + (size_t)l*H*H,     wl + OFF_WU,   H,   H);
        pack_w_t<<<(128*128 + 255)/256, 256, 0, stream>>>(
            WV    + (size_t)l*H*H,     wl + OFF_WV,   H,   H);
        pack_w_t<<<(256*128 + 255)/256, 256, 0, stream>>>(
            W1u   + (size_t)l*2*H*H,   wl + OFF_W1U,  2*H, H);
        pack_w_t<<<(128*384 + 255)/256, 256, 0, stream>>>(
            W2u   + (size_t)l*H*3*H,   wl + OFF_W2U,  H,   3*H);
    }

    // ---------------- embedding + edge geometry ----------------
    embed_kernel<<<n, 128, 0, stream>>>(node_attr, x, Wv_embed, W1e, b1e, W2e, b2e,
                                        s, v, n);
    edge_pre<<<(E + 127) / 128, 128, 0, stream>>>(edge_attr, dij, dire, E);

    const dim3 blk(128);
    const dim3 gN_N128((n + 63) / 64, 2);         // (n, *, 128)
    const dim3 gN_N384((n + 63) / 64, 6);         // (n, *, 384)
    const dim3 g2N_N128((2 * n + 63) / 64, 2);    // (2n, 128, 128)
    const long sN   = (long)n * 128;
    const long vN   = (long)n * 256;

    for (int l = 0; l < 4; ++l) {
        const __bf16* wl = wbf + (size_t)l * LSTRIDE;
        const float* bphi1_l = bphi1 + (size_t)l * H;
        const float* bphi2_l = bphi2 + (size_t)l * 3 * H;
        const float* Wfilt_l = Wfilt + (size_t)l * 3 * H;
        const float* bfilt_l = bfilt + (size_t)l * 3 * H;
        const float* b1u_l   = b1u   + (size_t)l * H;
        const float* b2u_l   = b2u   + (size_t)l * 3 * H;

        // phi = silu(s @ Wphi1 + b) @ Wphi2 + b
        pack_bf16<<<(sN + 255) / 256, 256, 0, stream>>>(s, sbf, sN);
        gemm_wmma_bf16<<<gN_N128, blk, 0, stream>>>(sbf, wl + OFF_PHI1, bphi1_l,
                                                    nullptr, hidbf, n, H, H, 1);
        gemm_wmma_bf16<<<gN_N384, blk, 0, stream>>>(hidbf, wl + OFF_PHI2, bphi2_l,
                                                    big, nullptr, n, H, 3 * H, 0);

        // snapshot v, then edge messages + scatter-add into s, v
        hipMemcpyAsync(vold, v, (size_t)n * 256 * sizeof(float),
                       hipMemcpyDeviceToDevice, stream);
        edge_msg<<<E, blk, 0, stream>>>(big, Wfilt_l, bfilt_l, dij, dire, vold,
                                        ej, ei, s, v, E);

        // Uv = v @ WU ; Vv2 = v @ WV  (v viewed as (2n, 128))
        pack_bf16<<<(vN + 255) / 256, 256, 0, stream>>>(v, vbf, vN);
        gemm_wmma_bf16<<<g2N_N128, blk, 0, stream>>>(vbf, wl + OFF_WU, nullptr,
                                                     U, nullptr, 2 * n, H, H, 0);
        gemm_wmma_bf16<<<g2N_N128, blk, 0, stream>>>(vbf, wl + OFF_WV, nullptr,
                                                     V2, nullptr, 2 * n, H, H, 0);

        // a = silu([s, ||Vv2||] @ W1u + b) @ W2u + b ; gated update
        ctx_kernel<<<n, blk, 0, stream>>>(s, V2, ctxbf, n);
        gemm_wmma_bf16<<<gN_N128, blk, 0, stream>>>(ctxbf, wl + OFF_W1U, b1u_l,
                                                    nullptr, hidbf, n, 2 * H, H, 1);
        gemm_wmma_bf16<<<gN_N384, blk, 0, stream>>>(hidbf, wl + OFF_W2U, b2u_l,
                                                    big, nullptr, n, H, 3 * H, 0);
        update_kernel<<<n, blk, 0, stream>>>(big, U, V2, s, v, n);
    }

    head_kernel<<<1, blk, 0, stream>>>(s, Wout, bout, Wfin, bfin, (float*)d_out);
}
